// Block_39926015984134
// MI455X (gfx1250) — matrix operand order repaired and checked by reading, not compiled, not run on hardware
//
#include <hip/hip_runtime.h>
#include <math.h>

typedef __attribute__((ext_vector_type(16))) __bf16 v16bf;
typedef __attribute__((ext_vector_type(8)))  float  v8f;

union AFrag { v16bf v; unsigned int u[8]; };
union BFrag { v16bf v; uint4 q[2]; };

#define BATCH 8
#define SEQ   2048
#define DIMSZ 512
#define HID   2048
#define NROWS (BATCH * SEQ)   // 16384

// A-fragment VGPR j <-> K-pair base (16-bit A 16x32 layout)
__device__ __forceinline__ int a_kpair(int j, int hi) {
  return ((j < 4) ? (2 * j) : (2 * j + 8)) + hi * 8;
}

// CDNA5 async DMA: global (16B) -> LDS, tracked by ASYNCcnt, no VGPR data.
__device__ __forceinline__ void async_ld16(unsigned int lds_off, unsigned long long gaddr) {
  asm volatile("global_load_async_to_lds_b128 %0, %1, off"
               :: "v"(lds_off), "v"(gaddr) : "memory");
}
__device__ __forceinline__ void wait_async0() {
  asm volatile("s_wait_asynccnt 0" ::: "memory");
}

// ---------------------------------------------------------------------------
// Elementwise f32 -> bf16 cast (weights)
// ---------------------------------------------------------------------------
__global__ __launch_bounds__(256)
void cast_bf16_kernel(const float* __restrict__ in, __bf16* __restrict__ out, int n) {
  int i = blockIdx.x * 256 + threadIdx.x;
  int stride = gridDim.x * 256;
  for (; i < n; i += stride) out[i] = (__bf16)in[i];
}

// ---------------------------------------------------------------------------
// Row LayerNorm over D=512 + affine + cast to bf16.  One block (256 thr) per row.
// ---------------------------------------------------------------------------
__global__ __launch_bounds__(256)
void ln_cast_kernel(const float* __restrict__ x, const float* __restrict__ g,
                    const float* __restrict__ b, __bf16* __restrict__ out) {
  __shared__ float s1[256], s2[256];
  const int row = blockIdx.x, t = threadIdx.x;
  const float* xr = x + (size_t)row * DIMSZ;
  float a0 = xr[t], a1 = xr[t + 256];
  s1[t] = a0 + a1;
  s2[t] = a0 * a0 + a1 * a1;
  __syncthreads();
  for (int off = 128; off > 0; off >>= 1) {
    if (t < off) { s1[t] += s1[t + off]; s2[t] += s2[t + off]; }
    __syncthreads();
  }
  float mu  = s1[0] * (1.0f / DIMSZ);
  float var = s2[0] * (1.0f / DIMSZ) - mu * mu;
  float rs  = rsqrtf(var + 1e-5f);
  __bf16* orow = out + (size_t)row * DIMSZ;
  orow[t]       = (__bf16)((a0 - mu) * rs * g[t]       + b[t]);
  orow[t + 256] = (__bf16)((a1 - mu) * rs * g[t + 256] + b[t + 256]);
}

// ---------------------------------------------------------------------------
// Tiled WMMA GEMM: C[M,N] = A[M,K] (bf16, row-major) x B[K,N] (bf16, row-major)
// Block tile 128x128, 8 waves each 32x64, K-step 32.
// Double-buffered LDS panels filled by global_load_async_to_lds_b128 so the
// next panel's DMA overlaps the current panel's WMMAs.
// ---------------------------------------------------------------------------
template<bool OUT_BF16, bool GELU_ACT, bool RESID>
__global__ __launch_bounds__(256)
void gemm_wmma_kernel(const __bf16* __restrict__ A, const __bf16* __restrict__ B,
                      int M, int N, int K,
                      const float* __restrict__ resid,
                      float* __restrict__ outF, __bf16* __restrict__ outB) {
  __shared__ __align__(16) __bf16 As[2][128 * 32];
  __shared__ __align__(16) __bf16 Bs[2][32 * 128];
  const int t = threadIdx.x;
  const int lane = t & 31, wave = t >> 5;
  const int wm = wave >> 1, wn = wave & 1;      // 4x2 wave grid over 128x128
  const int m0 = blockIdx.y * 128, n0 = blockIdx.x * 128;
  const int hi = lane >> 4, lm = lane & 15;

  // per-thread staging granules (2 for A, 2 for B per panel)
  const int ra0 = t >> 2,        ca0 = (t & 3) << 3;           // idx = t
  const int ra1 = (t + 256) >> 2, ca1 = ((t + 256) & 3) << 3;  // idx = t+256
  const int rb0 = t >> 4,        cb0 = (t & 15) << 3;
  const int rb1 = (t + 256) >> 4, cb1 = ((t + 256) & 15) << 3;

  auto stage = [&](int buf, int k0) {
    async_ld16((unsigned int)(uintptr_t)(&As[buf][ra0 * 32 + ca0]),
               (unsigned long long)(uintptr_t)(A + (size_t)(m0 + ra0) * K + (k0 + ca0)));
    async_ld16((unsigned int)(uintptr_t)(&As[buf][ra1 * 32 + ca1]),
               (unsigned long long)(uintptr_t)(A + (size_t)(m0 + ra1) * K + (k0 + ca1)));
    async_ld16((unsigned int)(uintptr_t)(&Bs[buf][rb0 * 128 + cb0]),
               (unsigned long long)(uintptr_t)(B + (size_t)(k0 + rb0) * N + (n0 + cb0)));
    async_ld16((unsigned int)(uintptr_t)(&Bs[buf][rb1 * 128 + cb1]),
               (unsigned long long)(uintptr_t)(B + (size_t)(k0 + rb1) * N + (n0 + cb1)));
  };

  v8f acc[2][4];
#pragma unroll
  for (int mi = 0; mi < 2; ++mi)
#pragma unroll
    for (int ni = 0; ni < 4; ++ni) acc[mi][ni] = {};

  stage(0, 0);
  wait_async0();
  __syncthreads();

  int cur = 0;
  for (int k0 = 0; k0 < K; k0 += 32) {
    const int nxt = cur ^ 1;
    if (k0 + 32 < K) stage(nxt, k0 + 32);   // DMA next panel during compute

    const __bf16* Ac = As[cur];
    const __bf16* Bc = Bs[cur];
    AFrag af[2];
#pragma unroll
    for (int mi = 0; mi < 2; ++mi) {
      int m = wm * 32 + mi * 16 + lm;
#pragma unroll
      for (int j = 0; j < 8; ++j)
        af[mi].u[j] = *(const unsigned int*)(Ac + m * 32 + a_kpair(j, hi));
    }
#pragma unroll
    for (int ni = 0; ni < 4; ++ni) {
      BFrag bf;
      int n = wn * 64 + ni * 16;
#pragma unroll
      for (int s = 0; s < 16; ++s)      // B 32x16: column = lane&15, K in table order
        bf.v[s] = Bc[(((s < 8) ? s : s + 8) + hi * 8) * 128 + n + lm];
#pragma unroll
      for (int mi = 0; mi < 2; ++mi)
        acc[mi][ni] = __builtin_amdgcn_wmma_f32_16x16x32_bf16(
            false, af[mi].v, false, bf.v, (short)0, acc[mi][ni], false, false);
    }
    wait_async0();
    __syncthreads();
    cur = nxt;
  }

  // Epilogue: C/D f32 layout -> m = j + 8*hi, n = lane&15
#pragma unroll
  for (int mi = 0; mi < 2; ++mi) {
#pragma unroll
    for (int ni = 0; ni < 4; ++ni) {
#pragma unroll
      for (int j = 0; j < 8; ++j) {
        int row = m0 + wm * 32 + mi * 16 + j + (hi << 3);
        int col = n0 + wn * 64 + ni * 16 + lm;
        size_t o = (size_t)row * N + col;
        float v = acc[mi][ni][j];
        if (GELU_ACT) v = 0.5f * v * (1.0f + erff(v * 0.70710678118654752f));
        if (RESID)          outF[o] = resid[o] + v;
        else if (OUT_BF16)  outB[o] = (__bf16)v;
        else                outF[o] = v;
      }
    }
  }
}

// ---------------------------------------------------------------------------
// Attention: one workgroup (4 waves, 128 thr) per (batch, 16-query tile).
// Pass A: scores[16,2048] = (Q Kt)/D via WMMA, Kt panels staged transposed in LDS.
// Softmax in LDS -> normalized P written as bf16.  Pass B: O = P V via WMMA.
// LDS: sc[16][2048] f32 (128K) | aux 64K (passA: Q 16K + Kt panel ~16.5K;
//      passB: P bf16 64K) | red[128] f32.
// ---------------------------------------------------------------------------
#define KT_STRIDE 264   // 256 keys + 8 halves pad (keeps 16B alignment)

__global__ __launch_bounds__(128)
void attn_kernel(const __bf16* __restrict__ qbf, const __bf16* __restrict__ kbf,
                 const __bf16* __restrict__ vbf, __bf16* __restrict__ obf) {
  extern __shared__ char smem_raw[];
  float*  sc  = (float*)smem_raw;                       // [16][2048]
  char*   aux = smem_raw + 16 * 2048 * 4;               // 64 KB region
  __bf16* qs  = (__bf16*)aux;                           // [16][512]   (pass A)
  __bf16* kT  = (__bf16*)(aux + 16 * 512 * 2);          // [32][264]   (pass A)
  __bf16* pbf = (__bf16*)aux;                           // [16][2048]  (pass B)
  float*  red = (float*)(smem_raw + 16 * 2048 * 4 + 65536);  // [128]

  const int t = threadIdx.x, lane = t & 31, wave = t >> 5;
  const int hi = lane >> 4, lm = lane & 15;
  const int b = blockIdx.y, qt = blockIdx.x;
  const size_t qbase  = ((size_t)b * SEQ + (size_t)qt * 16) * DIMSZ;
  const size_t kvbase = (size_t)b * SEQ * DIMSZ;

  // stage Q tile (coalesced b128)
  for (int i4 = t; i4 < (16 * DIMSZ) / 8; i4 += 128)
    ((uint4*)qs)[i4] = ((const uint4*)(qbf + qbase))[i4];
  __syncthreads();

  // ---- Pass A: panels of 256 keys; wave w owns keys [w*64, w*64+64) in panel ----
  for (int p = 0; p < SEQ / 256; ++p) {
    v8f acc[4];
#pragma unroll
    for (int ni = 0; ni < 4; ++ni) acc[ni] = {};

    for (int dk = 0; dk < DIMSZ; dk += 32) {
      // stage transposed Kt[32][256]: coalesced b128 reads, b16 scattered writes
      for (int task = t; task < 1024; task += 128) {
        int j = task >> 2;            // key within panel
        int c = (task & 3) << 3;      // d offset within 32 (8 halves)
        uint4 val = *(const uint4*)(kbf + kvbase +
                                    (size_t)(p * 256 + j) * DIMSZ + dk + c);
        __bf16 tmp[8];
        *(uint4*)tmp = val;
#pragma unroll
        for (int q = 0; q < 8; ++q) kT[(c + q) * KT_STRIDE + j] = tmp[q];
      }
      __syncthreads();

      AFrag af;
#pragma unroll
      for (int j = 0; j < 8; ++j)
        af.u[j] = *(const unsigned int*)(qs + lm * DIMSZ + dk + a_kpair(j, hi));
#pragma unroll
      for (int ni = 0; ni < 4; ++ni) {
        BFrag bf;
        int n = wave * 64 + ni * 16;
#pragma unroll
        for (int s = 0; s < 16; ++s)
          bf.v[s] = kT[(((s < 8) ? s : s + 8) + hi * 8) * KT_STRIDE + n + lm];
        acc[ni] = __builtin_amdgcn_wmma_f32_16x16x32_bf16(
            false, af.v, false, bf.v, (short)0, acc[ni], false, false);
      }
      __syncthreads();
    }
#pragma unroll
    for (int ni = 0; ni < 4; ++ni)
#pragma unroll
      for (int j = 0; j < 8; ++j)
        sc[(j + (hi << 3)) * SEQ + p * 256 + wave * 64 + ni * 16 + lm] =
            acc[ni][j] * (1.0f / (float)DIMSZ);
  }
  __syncthreads();

  // ---- Softmax: 8 threads per row, 256 cols each; emit normalized P as bf16 ----
  const int r = t >> 3, sub = t & 7;
  float* srow = sc + r * SEQ;
  const int c0 = sub * 256;
  float lmax = -3.4e38f;
  for (int c = 0; c < 256; ++c) lmax = fmaxf(lmax, srow[c0 + c]);
  red[t] = lmax;
  __syncthreads();
  float rmax = red[r * 8];
#pragma unroll
  for (int i = 1; i < 8; ++i) rmax = fmaxf(rmax, red[r * 8 + i]);
  __syncthreads();
  float lsum = 0.0f;
  for (int c = 0; c < 256; ++c) {
    float e = __expf(srow[c0 + c] - rmax);
    srow[c0 + c] = e;
    lsum += e;
  }
  red[t] = lsum;
  __syncthreads();
  float rsum = 0.0f;
#pragma unroll
  for (int i = 0; i < 8; ++i) rsum += red[r * 8 + i];
  float inv = 1.0f / rsum;
  __bf16* prow = pbf + r * SEQ;     // overwrites dead Q/Kt region
  for (int c = 0; c < 256; ++c) prow[c0 + c] = (__bf16)(srow[c0 + c] * inv);
  __syncthreads();

  // ---- Pass B: O = P @ V, wave owns output dims [wave*128, wave*128+128) ----
  const int d0 = wave * 128;
  v8f oacc[8];
#pragma unroll
  for (int nt = 0; nt < 8; ++nt) oacc[nt] = {};
  for (int kk = 0; kk < SEQ; kk += 32) {
    AFrag af;
#pragma unroll
    for (int j = 0; j < 8; ++j)
      af.u[j] = *(const unsigned int*)(pbf + lm * SEQ + kk + a_kpair(j, hi));
#pragma unroll
    for (int nt = 0; nt < 8; ++nt) {
      BFrag bf;
      const __bf16* vr = vbf + kvbase + (size_t)kk * DIMSZ + d0 + nt * 16 + lm;
#pragma unroll
      for (int s = 0; s < 16; ++s)       // B 32x16: column = lane&15, K in table order
        bf.v[s] = vr[(size_t)(((s < 8) ? s : s + 8) + hi * 8) * DIMSZ];
      oacc[nt] = __builtin_amdgcn_wmma_f32_16x16x32_bf16(
          false, af.v, false, bf.v, (short)0, oacc[nt], false, false);
    }
  }
#pragma unroll
  for (int nt = 0; nt < 8; ++nt)
#pragma unroll
    for (int j = 0; j < 8; ++j) {
      int row = j + (hi << 3);
      int col = d0 + nt * 16 + lm;
      obf[qbase + (size_t)row * DIMSZ + col] = (__bf16)oacc[nt][j];
    }
}

// ---------------------------------------------------------------------------
// Launcher
// ---------------------------------------------------------------------------
extern "C" void kernel_launch(void* const* d_in, const int* in_sizes, int n_in,
                              void* d_out, int out_size, void* d_ws, size_t ws_size,
                              hipStream_t stream) {
  const float* x  = (const float*)d_in[0];
  const float* wq = (const float*)d_in[1];
  const float* wk = (const float*)d_in[2];
  const float* wv = (const float*)d_in[3];
  const float* wp = (const float*)d_in[4];
  const float* w1 = (const float*)d_in[5];
  const float* w2 = (const float*)d_in[6];
  const float* g1 = (const float*)d_in[7];
  const float* b1 = (const float*)d_in[8];
  const float* g2 = (const float*)d_in[9];
  const float* b2 = (const float*)d_in[10];
  float* xout = (float*)d_out;

  char* ws = (char*)d_ws;
  size_t off = 0;
  auto take = [&](size_t bytes) -> char* {
    char* p = ws + off;
    off += (bytes + 255) & ~(size_t)255;
    return p;
  };
  __bf16* hbf  = (__bf16*)take((size_t)NROWS * DIMSZ * 2);
  __bf16* qbf  = (__bf16*)take((size_t)NROWS * DIMSZ * 2);
  __bf16* kbf  = (__bf16*)take((size_t)NROWS * DIMSZ * 2);
  __bf16* vbf  = (__bf16*)take((size_t)NROWS * DIMSZ * 2);
  __bf16* obf  = (__bf16*)take((size_t)NROWS * DIMSZ * 2);
  __bf16* h2bf = (__bf16*)take((size_t)NROWS * DIMSZ * 2);
  __bf16* ubf  = (__bf16*)take((size_t)NROWS * HID * 2);
  __bf16* wqb  = (__bf16*)take((size_t)DIMSZ * DIMSZ * 2);
  __bf16* wkb  = (__bf16*)take((size_t)DIMSZ * DIMSZ * 2);
  __bf16* wvb  = (__bf16*)take((size_t)DIMSZ * DIMSZ * 2);
  __bf16* wpb  = (__bf16*)take((size_t)DIMSZ * DIMSZ * 2);
  __bf16* w1b  = (__bf16*)take((size_t)DIMSZ * HID * 2);
  __bf16* w2b  = (__bf16*)take((size_t)HID * DIMSZ * 2);

  // 1. weight casts
  cast_bf16_kernel<<<64, 256, 0, stream>>>(wq, wqb, DIMSZ * DIMSZ);
  cast_bf16_kernel<<<64, 256, 0, stream>>>(wk, wkb, DIMSZ * DIMSZ);
  cast_bf16_kernel<<<64, 256, 0, stream>>>(wv, wvb, DIMSZ * DIMSZ);
  cast_bf16_kernel<<<64, 256, 0, stream>>>(wp, wpb, DIMSZ * DIMSZ);
  cast_bf16_kernel<<<256, 256, 0, stream>>>(w1, w1b, DIMSZ * HID);
  cast_bf16_kernel<<<256, 256, 0, stream>>>(w2, w2b, HID * DIMSZ);

  // 2. LN1
  ln_cast_kernel<<<NROWS, 256, 0, stream>>>(x, g1, b1, hbf);

  // 3. Q, K, V GEMMs (bf16 out)
  dim3 gD(DIMSZ / 128, NROWS / 128);
  gemm_wmma_kernel<true, false, false><<<gD, 256, 0, stream>>>(hbf, wqb, NROWS, DIMSZ, DIMSZ, nullptr, nullptr, qbf);
  gemm_wmma_kernel<true, false, false><<<gD, 256, 0, stream>>>(hbf, wkb, NROWS, DIMSZ, DIMSZ, nullptr, nullptr, kbf);
  gemm_wmma_kernel<true, false, false><<<gD, 256, 0, stream>>>(hbf, wvb, NROWS, DIMSZ, DIMSZ, nullptr, nullptr, vbf);

  // 4. attention
  size_t smem = (size_t)16 * 2048 * 4 + 65536 + 128 * 4;   // 197120 B
  hipFuncSetAttribute(reinterpret_cast<const void*>(attn_kernel),
                      hipFuncAttributeMaxDynamicSharedMemorySize, (int)smem);
  attn_kernel<<<dim3(SEQ / 16, BATCH), 128, smem, stream>>>(qbf, kbf, vbf, obf);

  // 5. x1 = x + O @ wp  -> d_out (f32)
  gemm_wmma_kernel<false, false, true><<<gD, 256, 0, stream>>>(obf, wpb, NROWS, DIMSZ, DIMSZ, x, xout, nullptr);

  // 6. LN2 on x1
  ln_cast_kernel<<<NROWS, 256, 0, stream>>>(xout, g2, b2, h2bf);

  // 7. u = gelu(h2 @ w1) (bf16 out)
  gemm_wmma_kernel<true, true, false><<<dim3(HID / 128, NROWS / 128), 256, 0, stream>>>(
      h2bf, w1b, NROWS, HID, DIMSZ, nullptr, nullptr, ubf);

  // 8. out = x1 + u @ w2  (read+write d_out, each element owned by one thread)
  gemm_wmma_kernel<false, false, true><<<gD, 256, 0, stream>>>(ubf, w2b, NROWS, DIMSZ, HID, xout, xout, nullptr);
}
